// AttentionBlock_29918742184450
// MI455X (gfx1250) — compile-verified, matrix operand order checked by reading, not run on hardware
//
#include <hip/hip_runtime.h>
#include <hip/hip_bf16.h>

typedef _Float16 v16h __attribute__((ext_vector_type(16)));
typedef _Float16 v8h  __attribute__((ext_vector_type(8)));
typedef float    v8f  __attribute__((ext_vector_type(8)));
typedef int      i4g  __attribute__((vector_size(16)));  // gcc-style int4 (builtin pointee)

constexpr int kB   = 8;      // batch
constexpr int kC   = 256;    // channels
constexpr int kN   = 16384;  // H*W tokens
constexpr int kG   = 32;     // groups
constexpr int kCPG = 8;      // channels per group
constexpr int kNH  = 8;      // heads
constexpr int kHD  = 32;     // head dim
constexpr float kScale = 0.17677669529663687f; // 1/sqrt(32)

// CDNA5 async global->LDS path (ASYNCcnt), gated on toolchain support.
#if defined(__gfx1250__) && \
    __has_builtin(__builtin_amdgcn_global_load_async_to_lds_b128) && \
    __has_builtin(__builtin_amdgcn_s_wait_asynccnt)
#define HAVE_ASYNC_LDS 1
typedef __attribute__((address_space(1))) i4g* as1_i4p;
typedef __attribute__((address_space(3))) i4g* as3_i4p;
#else
#define HAVE_ASYNC_LDS 0
#endif

// ---- fragment helpers -------------------------------------------------------
// 16-bit A/B fragment for v_wmma_f32_16x16x32_f16 (wave32):
// lane L (half=L>>4, r=L&15) holds row r; K indices = kb+{0..7} (v0-3) and
// kb+16+{0..7} (v4-7) with kb = k0 + half*8.  Caller passes p = row_base + kb.
__device__ __forceinline__ v16h ldfrag(const _Float16* p) {
  v8h lo = *(const v8h*)(p);
  v8h hi = *(const v8h*)(p + 16);
  return __builtin_shufflevector(lo, hi, 0,1,2,3,4,5,6,7,8,9,10,11,12,13,14,15);
}
__device__ __forceinline__ v8f wmma16(v16h a, v16h b, v8f c) {
  // (neg_a, A, neg_b, B, c_mod, C, reuse_a, reuse_b)
  return __builtin_amdgcn_wmma_f32_16x16x32_f16(false, a, false, b, (short)0, c,
                                                false, false);
}

// ---- 0: weight conversion f32 -> f16 ---------------------------------------
__global__ void convert_weights(const float* __restrict__ qkv_w,
                                const float* __restrict__ proj_w,
                                _Float16* __restrict__ qw,
                                _Float16* __restrict__ pw) {
  int i = blockIdx.x * 256 + threadIdx.x;
  if (i < 3 * kC * kC) qw[i] = (_Float16)qkv_w[i];
  if (i < kC * kC)     pw[i] = (_Float16)proj_w[i];
}

// ---- 1: GroupNorm statistics (one block per (b,g)) -------------------------
__global__ void __launch_bounds__(256) gn_stats(const float* __restrict__ x,
                                                float* __restrict__ stats) {
  int bg = blockIdx.x;
  const float4* xp = (const float4*)(x + ((size_t)bg << 17)); // group contiguous
  float s = 0.f, s2 = 0.f;
  for (int i = threadIdx.x; i < (kCPG * kN) / 4; i += 256) {
    float4 v = xp[i];
    s  += v.x + v.y + v.z + v.w;
    s2 += v.x*v.x + v.y*v.y + v.z*v.z + v.w*v.w;
  }
#pragma unroll
  for (int off = 16; off; off >>= 1) {
    s  += __shfl_xor(s,  off, 32);
    s2 += __shfl_xor(s2, off, 32);
  }
  __shared__ float rs[8], rs2[8];
  int w = threadIdx.x >> 5, lane = threadIdx.x & 31;
  if (lane == 0) { rs[w] = s; rs2[w] = s2; }
  __syncthreads();
  if (threadIdx.x == 0) {
    float S = 0.f, S2 = 0.f;
#pragma unroll
    for (int i = 0; i < 8; i++) { S += rs[i]; S2 += rs2[i]; }
    const float inv = 1.0f / (float)(kCPG * kN);
    float mean = S * inv;
    float var  = S2 * inv - mean * mean;
    stats[bg * 2 + 0] = mean;
    stats[bg * 2 + 1] = rsqrtf(var + 1e-5f);
  }
}

// ---- 2: GroupNorm apply -> h_t[b][n][c] (f16, transposed for GEMM-B) -------
__global__ void __launch_bounds__(256) gn_apply(const float* __restrict__ x,
                                                const float* __restrict__ gn_w,
                                                const float* __restrict__ gn_b,
                                                const float* __restrict__ stats,
                                                _Float16* __restrict__ h_t) {
  int bg = blockIdx.x;
  int b = bg >> 5, g = bg & 31;
  float mean = stats[bg * 2 + 0], rstd = stats[bg * 2 + 1];
  float wv[kCPG], bv[kCPG];
#pragma unroll
  for (int cc = 0; cc < kCPG; cc++) {
    float wc = gn_w[g * kCPG + cc] * rstd;
    wv[cc] = wc;
    bv[cc] = gn_b[g * kCPG + cc] - mean * wc;
  }
  const float* xp = x + ((size_t)(b * kC + g * kCPG) << 14);
  _Float16* hp = h_t + (((size_t)b << 14) * kC) + g * kCPG;
  for (int n = threadIdx.x; n < kN; n += 256) {
    v8h o;
#pragma unroll
    for (int cc = 0; cc < kCPG; cc++)
      o[cc] = (_Float16)(xp[(size_t)cc * kN + n] * wv[cc] + bv[cc]);
    *(v8h*)(hp + (size_t)n * kC) = o;  // 16B store
  }
}

// ---- 3: QKV GEMM: (768x256) x h_t^T -> q,k (n-major) and v (d-major) -------
__global__ void __launch_bounds__(256) qkv_gemm(const _Float16* __restrict__ qw,
                                                const float* __restrict__ qkv_b,
                                                const _Float16* __restrict__ h_t,
                                                _Float16* __restrict__ qbuf,
                                                _Float16* __restrict__ kbuf,
                                                _Float16* __restrict__ vtbuf) {
  int b  = blockIdx.z;
  int o0 = blockIdx.y * 16;
  int wv = threadIdx.x >> 5;
  int n0 = blockIdx.x * 512 + wv * 64;
  int lane = threadIdx.x & 31, half = lane >> 4, r = lane & 15;
  const _Float16* hb = h_t + ((size_t)b << 14) * kC;
  v8f zf = {};
  v8f acc0 = zf, acc1 = zf, acc2 = zf, acc3 = zf;
#pragma unroll
  for (int k0 = 0; k0 < kC; k0 += 32) {
    int kb = k0 + half * 8;
    v16h a = ldfrag(qw + (size_t)(o0 + r) * kC + kb);
    acc0 = wmma16(a, ldfrag(hb + (size_t)(n0 +  0 + r) * kC + kb), acc0);
    acc1 = wmma16(a, ldfrag(hb + (size_t)(n0 + 16 + r) * kC + kb), acc1);
    acc2 = wmma16(a, ldfrag(hb + (size_t)(n0 + 32 + r) * kC + kb), acc2);
    acc3 = wmma16(a, ldfrag(hb + (size_t)(n0 + 48 + r) * kC + kb), acc3);
  }
  int part  = o0 >> 8;              // 0=q,1=k,2=v (uniform per block)
  int obase = o0 + half * 8;
  v8f accs[4] = {acc0, acc1, acc2, acc3};
#pragma unroll
  for (int i = 0; i < 8; i++) {
    int o = obase + i;
    int co = o & 255, head = co >> 5, dd = co & 31;
    float bia = qkv_b[o];
    size_t bh = (size_t)(b * kNH + head);
#pragma unroll
    for (int s = 0; s < 4; s++) {
      int n = n0 + s * 16 + r;
      float v = accs[s][i] + bia;
      if (part == 0) {
        qbuf[((bh << 14) + n) * kHD + dd] = (_Float16)(v * kScale);
      } else if (part == 1) {
        kbuf[((bh << 14) + n) * kHD + dd] = (_Float16)v;
      } else {
        vtbuf[(bh * kHD + dd) * (size_t)kN + n] = (_Float16)v;
      }
    }
  }
}

// ---- 4: flash attention per (b,head); wave = 16 query rows -----------------
// K/V tiles are staged in LDS once per block (double-buffered, async DMA on
// CDNA5: global_load_async_to_lds_b128 + s_wait_asynccnt).
__global__ void __launch_bounds__(256) attn_kernel(const _Float16* __restrict__ qbuf,
                                                   const _Float16* __restrict__ kbuf,
                                                   const _Float16* __restrict__ vtbuf,
                                                   _Float16* __restrict__ att_t) {
  int bh = blockIdx.y;                 // b*8 + head
  int b = bh >> 3, head = bh & 7;
  int wv = threadIdx.x >> 5;
  int tid = threadIdx.x;
  int lane = threadIdx.x & 31, half = lane >> 4, r = lane & 15;
  int q0 = blockIdx.x * 128 + wv * 16;

  const _Float16* qp = qbuf  + ((size_t)bh << 14) * kHD;   // [n][dd]
  const _Float16* kp = kbuf  + ((size_t)bh << 14) * kHD;   // [n][dd]
  const _Float16* vp = vtbuf + ((size_t)bh * kHD) * kN;    // [dd][n]

  // padded (stride 72 halves = 144B) tiles: even bank use for DS b128 reads
  __shared__ __align__(16) _Float16 ktile[2][64][72];  // [buf][key][dd]
  __shared__ __align__(16) _Float16 vtile[2][32][72];  // [buf][dd][key]
  __shared__ __align__(16) _Float16 pbuf[8][16][72];   // per-wave P tile

  int kb = half * 8;
  v16h qf = ldfrag(qp + (size_t)(q0 + r) * kHD + kb);  // A-frag, reused all tiles

  v8f zf = {};
  v8f oA = zf, oB = zf;                // O columns dd=r and dd=16+r
  float mrow[8], lrow[8];
#pragma unroll
  for (int i = 0; i < 8; i++) { mrow[i] = -1e30f; lrow[i] = 0.f; }

  // per-thread tile-copy addressing: 256 threads x 16B cover each 4KB tile
  int krow = tid >> 2, kseg = tid & 3;   // K: 64 rows x 4 chunks
  int vrow = tid >> 3, vseg = tid & 7;   // V: 32 rows x 8 chunks

  auto issue_copy = [&](int j0, int bufi) {
    const _Float16* gk = kp + (size_t)(j0 + krow) * kHD + kseg * 8;
    const _Float16* gv = vp + (size_t)vrow * kN + j0 + vseg * 8;
    _Float16* lk = &ktile[bufi][krow][kseg * 8];
    _Float16* lv = &vtile[bufi][vrow][vseg * 8];
#if HAVE_ASYNC_LDS
    __builtin_amdgcn_global_load_async_to_lds_b128(
        (as1_i4p)(void*)gk, (as3_i4p)(void*)lk, 0, 0);
    __builtin_amdgcn_global_load_async_to_lds_b128(
        (as1_i4p)(void*)gv, (as3_i4p)(void*)lv, 0, 0);
#else
    *(v8h*)lk = *(const v8h*)gk;
    *(v8h*)lv = *(const v8h*)gv;
    if (j0 + 64 < kN) {  // warm next tile (global_prefetch_b8)
      __builtin_prefetch(gk + 64 * kHD, 0, 1);
      __builtin_prefetch(gv + 64, 0, 1);
    }
#endif
  };

  int cur = 0;
  issue_copy(0, 0);
  for (int j0 = 0; j0 < kN; j0 += 64) {
    bool hasNext = (j0 + 64) < kN;
    __syncthreads();                   // everyone done reading buf[cur^1]
    if (hasNext) issue_copy(j0 + 64, cur ^ 1);
#if HAVE_ASYNC_LDS
    if (hasNext) __builtin_amdgcn_s_wait_asynccnt(2);
    else         __builtin_amdgcn_s_wait_asynccnt(0);
#endif
    __syncthreads();                   // buf[cur] visible to all waves

    // S = q . k^T   (4x 16x16 tiles over 64 keys) from LDS
    v8f s[4];
#pragma unroll
    for (int t = 0; t < 4; t++) {
      v16h kf = ldfrag(&ktile[cur][t * 16 + r][kb]);
      s[t] = wmma16(qf, kf, zf);
    }
    // online softmax: row stats live in lanes of the same half-wave
    float tmax[8];
#pragma unroll
    for (int i = 0; i < 8; i++)
      tmax[i] = fmaxf(fmaxf(s[0][i], s[1][i]), fmaxf(s[2][i], s[3][i]));
#pragma unroll
    for (int off = 1; off < 16; off <<= 1)
#pragma unroll
      for (int i = 0; i < 8; i++)
        tmax[i] = fmaxf(tmax[i], __shfl_xor(tmax[i], off, 32));
    float alpha[8], rsum[8];
#pragma unroll
    for (int i = 0; i < 8; i++) {
      float mn = fmaxf(mrow[i], tmax[i]);
      alpha[i] = __expf(mrow[i] - mn);
      mrow[i] = mn;
      rsum[i] = 0.f;
    }
#pragma unroll
    for (int t = 0; t < 4; t++)
#pragma unroll
      for (int i = 0; i < 8; i++) {
        float p = __expf(s[t][i] - mrow[i]);
        s[t][i] = p;
        rsum[i] += p;
      }
#pragma unroll
    for (int off = 1; off < 16; off <<= 1)
#pragma unroll
      for (int i = 0; i < 8; i++)
        rsum[i] += __shfl_xor(rsum[i], off, 32);
#pragma unroll
    for (int i = 0; i < 8; i++) {
      lrow[i] = lrow[i] * alpha[i] + rsum[i];
      oA[i] *= alpha[i];
      oB[i] *= alpha[i];
    }
    // P (C-layout) -> LDS -> A-layout fragments (wave-private region)
#pragma unroll
    for (int t = 0; t < 4; t++)
#pragma unroll
      for (int i = 0; i < 8; i++)
        pbuf[wv][half * 8 + i][t * 16 + r] = (_Float16)s[t][i];
    // O += P . V   (K-dim 64 = 2 chunks of 32; dd split in two 16-col tiles)
#pragma unroll
    for (int c = 0; c < 2; c++) {
      v16h pa  = ldfrag(&pbuf[wv][r][c * 32 + kb]);
      v16h vb0 = ldfrag(&vtile[cur][r][c * 32 + kb]);
      v16h vb1 = ldfrag(&vtile[cur][16 + r][c * 32 + kb]);
      oA = wmma16(pa, vb0, oA);
      oB = wmma16(pa, vb1, oB);
    }
    cur ^= 1;
  }
  // normalize + write transposed att_t[b][n][c] for the proj GEMM
  _Float16* ap = att_t + (((size_t)b << 14) * kC) + head * kHD;
#pragma unroll
  for (int i = 0; i < 8; i++) {
    float inv = 1.0f / lrow[i];
    size_t n = (size_t)(q0 + half * 8 + i);
    ap[n * kC + r]      = (_Float16)(oA[i] * inv);
    ap[n * kC + 16 + r] = (_Float16)(oB[i] * inv);
  }
}

// ---- 5: output projection + bias + residual --------------------------------
__global__ void __launch_bounds__(256) proj_kernel(const _Float16* __restrict__ pw,
                                                   const float* __restrict__ proj_b,
                                                   const _Float16* __restrict__ att_t,
                                                   const float* __restrict__ x,
                                                   float* __restrict__ out) {
  int b  = blockIdx.z;
  int o0 = blockIdx.y * 16;
  int wv = threadIdx.x >> 5;
  int n0 = blockIdx.x * 512 + wv * 64;
  int lane = threadIdx.x & 31, half = lane >> 4, r = lane & 15;
  const _Float16* ab = att_t + ((size_t)b << 14) * kC;
  v8f zf = {};
  v8f acc0 = zf, acc1 = zf, acc2 = zf, acc3 = zf;
#pragma unroll
  for (int k0 = 0; k0 < kC; k0 += 32) {
    int kb = k0 + half * 8;
    v16h a = ldfrag(pw + (size_t)(o0 + r) * kC + kb);
    acc0 = wmma16(a, ldfrag(ab + (size_t)(n0 +  0 + r) * kC + kb), acc0);
    acc1 = wmma16(a, ldfrag(ab + (size_t)(n0 + 16 + r) * kC + kb), acc1);
    acc2 = wmma16(a, ldfrag(ab + (size_t)(n0 + 32 + r) * kC + kb), acc2);
    acc3 = wmma16(a, ldfrag(ab + (size_t)(n0 + 48 + r) * kC + kb), acc3);
  }
  v8f accs[4] = {acc0, acc1, acc2, acc3};
#pragma unroll
  for (int i = 0; i < 8; i++) {
    int o = o0 + half * 8 + i;
    float bia = proj_b[o];
    const float* xr  = x   + ((size_t)(b * kC + o) << 14);
    float*       orw = out + ((size_t)(b * kC + o) << 14);
#pragma unroll
    for (int s = 0; s < 4; s++) {
      int n = n0 + s * 16 + r;
      orw[n] = xr[n] + accs[s][i] + bia;
    }
  }
}

// ---- launch ----------------------------------------------------------------
extern "C" void kernel_launch(void* const* d_in, const int* in_sizes, int n_in,
                              void* d_out, int out_size, void* d_ws, size_t ws_size,
                              hipStream_t stream) {
  const float* x      = (const float*)d_in[0];
  const float* gn_w   = (const float*)d_in[1];
  const float* gn_b   = (const float*)d_in[2];
  const float* qkv_w  = (const float*)d_in[3];
  const float* qkv_b  = (const float*)d_in[4];
  const float* proj_w = (const float*)d_in[5];
  const float* proj_b = (const float*)d_in[6];
  float* out = (float*)d_out;

  char* ws = (char*)d_ws;
  const size_t F16 = (size_t)kB * kN * kC * sizeof(_Float16); // 67,108,864 B
  _Float16* qw_h   = (_Float16*)(ws);                         // 768*256 f16
  _Float16* pw_h   = (_Float16*)(ws + 393216);                // 256*256 f16
  float*    stats  = (float*)(ws + 524288);                   // 512 f32
  _Float16* h_t    = (_Float16*)(ws + 526336);
  _Float16* qbuf   = (_Float16*)(ws + 526336 + F16);
  _Float16* kbuf   = (_Float16*)(ws + 526336 + 2 * F16);
  _Float16* vtbuf  = (_Float16*)(ws + 526336 + 3 * F16);
  _Float16* att_t  = (_Float16*)(ws + 526336 + 4 * F16);

  convert_weights<<<768, 256, 0, stream>>>(qkv_w, proj_w, qw_h, pw_h);
  gn_stats<<<kB * kG, 256, 0, stream>>>(x, stats);
  gn_apply<<<kB * kG, 256, 0, stream>>>(x, gn_w, gn_b, stats, h_t);
  qkv_gemm<<<dim3(kN / 512, (3 * kC) / 16, kB), 256, 0, stream>>>(
      qw_h, qkv_b, h_t, qbuf, kbuf, vtbuf);
  attn_kernel<<<dim3(kN / 128, kB * kNH), 256, 0, stream>>>(
      qbuf, kbuf, vtbuf, att_t);
  proj_kernel<<<dim3(kN / 512, kC / 16, kB), 256, 0, stream>>>(
      pw_h, proj_b, att_t, x, out);
}
